// RBF_67577015435739
// MI455X (gfx1250) — matrix operand order repaired and checked by reading, not compiled, not run on hardware
//
#include <hip/hip_runtime.h>
#include <hip/hip_bf16.h>

// ---------------------------------------------------------------------------
// RBF layer: out[b,c] = exp(-(||x_b||^2 + ||c_c||^2 - 2 x_b . c_c) / (2 s_c^2))
// B=8192, C=2048, F=512, f32 in/out.
// Cross-term GEMM via V_WMMA_F32_16X16X32_BF16 with 3-product bf16 hi/lo
// split (hi*hi + hi*lo + lo*hi), f32 accumulation.
// Fast path: pre-split bf16 operands in workspace + double-buffered LDS with
// GLOBAL_LOAD_ASYNC_TO_LDS_B128 prefetch (ASYNCcnt) when available.
// ---------------------------------------------------------------------------

#define BATCH 8192
#define NCENT 2048
#define NFEAT 512

typedef __bf16 bf16_t;
typedef bf16_t v16bf __attribute__((ext_vector_type(16)));
typedef float  v8f   __attribute__((ext_vector_type(8)));
typedef int    v4i_vec __attribute__((vector_size(16)));

union FragU {
    v16bf v;
    uint4 q[2];
};

#ifndef __has_builtin
#define __has_builtin(x) 0
#endif
#if __has_builtin(__builtin_amdgcn_global_load_async_to_lds_b128) && \
    __has_builtin(__builtin_amdgcn_s_wait_asynccnt)
#define USE_ASYNC_LDS 1
#else
#define USE_ASYNC_LDS 0
#endif

// round-to-nearest-even f32 -> bf16 bits
__device__ __forceinline__ unsigned int f2bf_bits(float f) {
    unsigned int u = __float_as_uint(f);
    u += 0x7fffu + ((u >> 16) & 1u);
    return u >> 16;
}
__device__ __forceinline__ float bf_bits2f(unsigned int b) {
    return __uint_as_float(b << 16);
}

// LDS tile: 128 rows x 32 k, padded stride 40 halves (80B): keeps per-lane
// 16B chunks 16B-aligned and spreads banks.
#define LDT 40
#define TILE_E (128 * LDT)   // ushort elements per tile

// ---------------------------------------------------------------------------
// Fragment loads from LDS (bf16 row-major [128][LDT] tiles)
// ---------------------------------------------------------------------------
// A (16x32, MxK): lane l: row (l&15); chunks at koff and koff+16, koff=(l<16)?0:8
__device__ __forceinline__ v16bf load_frag_A(const unsigned short* __restrict__ lds,
                                             int tile_row, int lane) {
    const int r    = tile_row + (lane & 15);
    const int koff = (lane < 16) ? 0 : 8;
    FragU f;
    f.q[0] = *(const uint4*)(lds + r * LDT + koff);
    f.q[1] = *(const uint4*)(lds + r * LDT + koff + 16);
    return f.v;
}
// B (32x16, KxN): lane l: column (l&15); chunks at koff and koff+16, koff=(l<16)?0:8
__device__ __forceinline__ v16bf load_frag_B(const unsigned short* __restrict__ lds,
                                             int tile_row, int lane) {
    const int r    = tile_row + (lane & 15);
    const int koff = (lane < 16) ? 0 : 8;
    FragU f;
    f.q[0] = *(const uint4*)(lds + r * LDT + koff);
    f.q[1] = *(const uint4*)(lds + r * LDT + koff + 16);
    return f.v;
}

__device__ __forceinline__ v8f wmma_bf16(v16bf a, v16bf b, v8f c) {
    return __builtin_amdgcn_wmma_f32_16x16x32_bf16(
        false, a, false, b, (short)0, c, false, false);
}

// 24 WMMAs on one staged k-chunk. base points at {Ahi,Alo,Bhi,Blo} tiles.
__device__ __forceinline__ void compute_chunk(const unsigned short* __restrict__ base,
                                              int wm, int wn, int lane,
                                              v8f (&acc)[2][4]) {
    const unsigned short* Ahi = base;
    const unsigned short* Alo = base + TILE_E;
    const unsigned short* Bhi = base + 2 * TILE_E;
    const unsigned short* Blo = base + 3 * TILE_E;

    v16bf ah[2], al[2], bh[4], bl[4];
#pragma unroll
    for (int i = 0; i < 2; ++i) {
        ah[i] = load_frag_A(Ahi, wm * 32 + i * 16, lane);
        al[i] = load_frag_A(Alo, wm * 32 + i * 16, lane);
    }
#pragma unroll
    for (int j = 0; j < 4; ++j) {
        bh[j] = load_frag_B(Bhi, wn * 64 + j * 16, lane);
        bl[j] = load_frag_B(Blo, wn * 64 + j * 16, lane);
    }
#pragma unroll
    for (int i = 0; i < 2; ++i)
#pragma unroll
        for (int j = 0; j < 4; ++j) {
            acc[i][j] = wmma_bf16(ah[i], bh[j], acc[i][j]);  // hi*hi
            acc[i][j] = wmma_bf16(ah[i], bl[j], acc[i][j]);  // hi*lo
            acc[i][j] = wmma_bf16(al[i], bh[j], acc[i][j]);  // lo*hi
        }
}

// ---------------------------------------------------------------------------
// RBF epilogue shared by both main kernels
// ---------------------------------------------------------------------------
__device__ __forceinline__ void rbf_epilogue(v8f (&acc)[2][4], int mbase, int nbase,
                                             int wm, int wn, int lane,
                                             const float* __restrict__ xsq,
                                             const float* __restrict__ csq,
                                             const float* __restrict__ inv2s2,
                                             float* __restrict__ out) {
    const int mhalf = (lane >> 4) & 1;   // C/D layout: lanes 16-31 -> +8 rows
#pragma unroll
    for (int j = 0; j < 4; ++j) {
        const int n  = nbase + wn * 64 + j * 16 + (lane & 15);
        const float cs = csq[n];
        const float iv = inv2s2[n];
#pragma unroll
        for (int i = 0; i < 2; ++i) {
            const int m0 = mbase + wm * 32 + i * 16 + mhalf * 8;
#pragma unroll
            for (int r = 0; r < 8; ++r) {
                const int m = m0 + r;
                float d = xsq[m] + cs - 2.0f * acc[i][j][r];
                d = fmaxf(d, 0.0f);
                out[(size_t)m * NCENT + n] = expf(-d * iv);
            }
        }
    }
}

// ---------------------------------------------------------------------------
// Row squared-norms (one wave32 per row); optional 1/(2 sigma^2).
// ---------------------------------------------------------------------------
__global__ void rbf_norms_kernel(const float* __restrict__ x, int nrows,
                                 float* __restrict__ sq_out,
                                 const float* __restrict__ sigmas,
                                 float* __restrict__ inv_out) {
    const int gw   = (blockIdx.x * blockDim.x + threadIdx.x) >> 5;
    const int lane = threadIdx.x & 31;
    if (gw >= nrows) return;
    const float* row = x + (size_t)gw * NFEAT;
    float s = 0.0f;
#pragma unroll
    for (int i = 0; i < NFEAT / 128; ++i) {
        float4 f = *(const float4*)(row + lane * 4 + i * 128);
        s += f.x * f.x + f.y * f.y + f.z * f.z + f.w * f.w;
    }
#pragma unroll
    for (int off = 16; off > 0; off >>= 1) s += __shfl_xor(s, off, 32);
    if (lane == 0) {
        sq_out[gw] = s;
        if (inv_out) {
            float sg = sigmas[gw];
            inv_out[gw] = 1.0f / (2.0f * sg * sg);
        }
    }
}

// ---------------------------------------------------------------------------
// Pre-split: f32 -> bf16 hi/lo arrays (RNE), 1 float4 per thread.
// ---------------------------------------------------------------------------
__global__ void rbf_split_kernel(const float* __restrict__ src,
                                 unsigned short* __restrict__ hi,
                                 unsigned short* __restrict__ lo, int n4) {
    const int i = blockIdx.x * blockDim.x + threadIdx.x;
    if (i >= n4) return;
    float4 t = ((const float4*)src)[i];
    unsigned int hx = f2bf_bits(t.x), hy = f2bf_bits(t.y);
    unsigned int hz = f2bf_bits(t.z), hw = f2bf_bits(t.w);
    float lx = t.x - bf_bits2f(hx), ly = t.y - bf_bits2f(hy);
    float lz = t.z - bf_bits2f(hz), lw = t.w - bf_bits2f(hw);
    ((uint2*)hi)[i] = make_uint2(hx | (hy << 16), hz | (hw << 16));
    ((uint2*)lo)[i] = make_uint2(f2bf_bits(lx) | (f2bf_bits(ly) << 16),
                                 f2bf_bits(lz) | (f2bf_bits(lw) << 16));
}

// ---------------------------------------------------------------------------
// Fast main kernel: pre-split bf16 operands, double-buffered LDS.
// Block 128x128, 256 threads = 8 waves (4 M x 2 N), each wave 32x64.
// ---------------------------------------------------------------------------
#if USE_ASYNC_LDS
__device__ __forceinline__ void async_copy32B(const unsigned short* g,
                                              unsigned short* l) {
    // Builtin signature (from clang diagnostic): param0 = addrspace(1) v4i*,
    // i.e. global pointer to a 16B int vector; param1 = LDS side.
    // INST_OFFSET applies to both global and LDS addresses (ISA 10.x).
    __attribute__((address_space(1))) v4i_vec* gp =
        (__attribute__((address_space(1))) v4i_vec*)(unsigned long long)g;
    __attribute__((address_space(3))) v4i_vec* lp =
        (__attribute__((address_space(3))) v4i_vec*)(unsigned int)(unsigned long long)l;
    __builtin_amdgcn_global_load_async_to_lds_b128(gp, lp, 0, 0);
    __builtin_amdgcn_global_load_async_to_lds_b128(gp, lp, 16, 0);
}
#endif

__global__ void __launch_bounds__(256)
rbf_main_split(const unsigned short* __restrict__ xhi,
               const unsigned short* __restrict__ xlo,
               const unsigned short* __restrict__ chi,
               const unsigned short* __restrict__ clo,
               const float* __restrict__ xsq,
               const float* __restrict__ csq,
               const float* __restrict__ inv2s2,
               float* __restrict__ out) {
    extern __shared__ __align__(16) unsigned char smem_raw[];
    unsigned short* smem = (unsigned short*)smem_raw;   // 2 bufs x 4 tiles

    const int tid  = threadIdx.x;
    const int lane = tid & 31;
    const int wave = tid >> 5;
    const int wm   = wave & 3;
    const int wn   = wave >> 2;
    const int mbase = blockIdx.y * 128;
    const int nbase = blockIdx.x * 128;

    // staging assignment: thread -> (row, 16-element half)
    const int row  = tid >> 1;
    const int half = tid & 1;
    const size_t gA = (size_t)(mbase + row) * NFEAT + half * 16;
    const size_t gB = (size_t)(nbase + row) * NFEAT + half * 16;
    const int loff  = row * LDT + half * 16;

    v8f acc[2][4];
#pragma unroll
    for (int i = 0; i < 2; ++i)
#pragma unroll
        for (int j = 0; j < 4; ++j) acc[i][j] = (v8f){0, 0, 0, 0, 0, 0, 0, 0};

#if USE_ASYNC_LDS
    // ---- async double-buffered pipeline ----
    {
        unsigned short* b0 = smem;
        async_copy32B(xhi + gA, b0 + 0 * TILE_E + loff);
        async_copy32B(xlo + gA, b0 + 1 * TILE_E + loff);
        async_copy32B(chi + gB, b0 + 2 * TILE_E + loff);
        async_copy32B(clo + gB, b0 + 3 * TILE_E + loff);
    }
    __builtin_amdgcn_s_wait_asynccnt(0);
    __syncthreads();

    int buf = 0;
    for (int kk = 0; kk < NFEAT; kk += 32) {
        const bool has_next = (kk + 32) < NFEAT;
        if (has_next) {
            unsigned short* nb = smem + (buf ^ 1) * 4 * TILE_E;
            const int kn = kk + 32;
            async_copy32B(xhi + gA + kn, nb + 0 * TILE_E + loff);
            async_copy32B(xlo + gA + kn, nb + 1 * TILE_E + loff);
            async_copy32B(chi + gB + kn, nb + 2 * TILE_E + loff);
            async_copy32B(clo + gB + kn, nb + 3 * TILE_E + loff);
        }
        compute_chunk(smem + buf * 4 * TILE_E, wm, wn, lane, acc);
        if (has_next) {
            __builtin_amdgcn_s_wait_asynccnt(0);
            __syncthreads();
        }
        buf ^= 1;
    }
#else
    // ---- register-prefetch double-buffered pipeline ----
    uint4 r[8];
#define LOAD_REGS(kkv)                                                        \
    do {                                                                      \
        r[0] = *(const uint4*)(xhi + gA + (kkv));                             \
        r[1] = *(const uint4*)(xhi + gA + (kkv) + 8);                         \
        r[2] = *(const uint4*)(xlo + gA + (kkv));                             \
        r[3] = *(const uint4*)(xlo + gA + (kkv) + 8);                         \
        r[4] = *(const uint4*)(chi + gB + (kkv));                             \
        r[5] = *(const uint4*)(chi + gB + (kkv) + 8);                         \
        r[6] = *(const uint4*)(clo + gB + (kkv));                             \
        r[7] = *(const uint4*)(clo + gB + (kkv) + 8);                         \
    } while (0)

    LOAD_REGS(0);
    int buf = 0;
    for (int kk = 0; kk < NFEAT; kk += 32) {
        unsigned short* b = smem + buf * 4 * TILE_E;
        ((uint4*)(b + 0 * TILE_E + loff))[0] = r[0];
        ((uint4*)(b + 0 * TILE_E + loff))[1] = r[1];
        ((uint4*)(b + 1 * TILE_E + loff))[0] = r[2];
        ((uint4*)(b + 1 * TILE_E + loff))[1] = r[3];
        ((uint4*)(b + 2 * TILE_E + loff))[0] = r[4];
        ((uint4*)(b + 2 * TILE_E + loff))[1] = r[5];
        ((uint4*)(b + 3 * TILE_E + loff))[0] = r[6];
        ((uint4*)(b + 3 * TILE_E + loff))[1] = r[7];
        if (kk + 32 < NFEAT) LOAD_REGS(kk + 32);
        __syncthreads();
        compute_chunk(b, wm, wn, lane, acc);
        buf ^= 1;
    }
#undef LOAD_REGS
#endif

    rbf_epilogue(acc, mbase, nbase, wm, wn, lane, xsq, csq, inv2s2, out);
}

// ---------------------------------------------------------------------------
// Fallback main kernel (workspace too small): convert f32 -> hi/lo in-kernel.
// ---------------------------------------------------------------------------
__device__ __forceinline__ void stage_tile_f32(const float* __restrict__ gsrc,
                                               int row_base, int kk,
                                               unsigned short* __restrict__ lds_hi,
                                               unsigned short* __restrict__ lds_lo,
                                               int tid) {
    const int row  = tid >> 1;
    const int half = tid & 1;
    const float* p = gsrc + (size_t)(row_base + row) * NFEAT + kk + half * 16;

    unsigned int hi[8], lo[8];
#pragma unroll
    for (int q = 0; q < 4; ++q) {
        float4 t = ((const float4*)p)[q];
        unsigned int hx = f2bf_bits(t.x), hy = f2bf_bits(t.y);
        unsigned int hz = f2bf_bits(t.z), hw = f2bf_bits(t.w);
        float lx = t.x - bf_bits2f(hx), ly = t.y - bf_bits2f(hy);
        float lz = t.z - bf_bits2f(hz), lw = t.w - bf_bits2f(hw);
        hi[2 * q + 0] = hx | (hy << 16);
        hi[2 * q + 1] = hz | (hw << 16);
        lo[2 * q + 0] = f2bf_bits(lx) | (f2bf_bits(ly) << 16);
        lo[2 * q + 1] = f2bf_bits(lz) | (f2bf_bits(lw) << 16);
    }
    const int off = row * LDT + half * 16;
    ((uint4*)(lds_hi + off))[0] = make_uint4(hi[0], hi[1], hi[2], hi[3]);
    ((uint4*)(lds_hi + off))[1] = make_uint4(hi[4], hi[5], hi[6], hi[7]);
    ((uint4*)(lds_lo + off))[0] = make_uint4(lo[0], lo[1], lo[2], lo[3]);
    ((uint4*)(lds_lo + off))[1] = make_uint4(lo[4], lo[5], lo[6], lo[7]);
}

__global__ void __launch_bounds__(256)
rbf_main_f32(const float* __restrict__ inputs,
             const float* __restrict__ centers,
             const float* __restrict__ xsq,
             const float* __restrict__ csq,
             const float* __restrict__ inv2s2,
             float* __restrict__ out) {
    __shared__ unsigned short sT[4 * TILE_E];   // Ahi, Alo, Bhi, Blo

    const int tid  = threadIdx.x;
    const int lane = tid & 31;
    const int wave = tid >> 5;
    const int wm   = wave & 3;
    const int wn   = wave >> 2;
    const int mbase = blockIdx.y * 128;
    const int nbase = blockIdx.x * 128;

    v8f acc[2][4];
#pragma unroll
    for (int i = 0; i < 2; ++i)
#pragma unroll
        for (int j = 0; j < 4; ++j) acc[i][j] = (v8f){0, 0, 0, 0, 0, 0, 0, 0};

    for (int kk = 0; kk < NFEAT; kk += 32) {
        if (kk) __syncthreads();
        stage_tile_f32(inputs,  mbase, kk, sT + 0 * TILE_E, sT + 1 * TILE_E, tid);
        stage_tile_f32(centers, nbase, kk, sT + 2 * TILE_E, sT + 3 * TILE_E, tid);
        __syncthreads();
        compute_chunk(sT, wm, wn, lane, acc);
    }
    rbf_epilogue(acc, mbase, nbase, wm, wn, lane, xsq, csq, inv2s2, out);
}

// ---------------------------------------------------------------------------
extern "C" void kernel_launch(void* const* d_in, const int* in_sizes, int n_in,
                              void* d_out, int out_size, void* d_ws, size_t ws_size,
                              hipStream_t stream) {
    (void)in_sizes; (void)n_in; (void)out_size;

    const float* inputs  = (const float*)d_in[0];   // [8192, 512]
    const float* centers = (const float*)d_in[1];   // [2048, 512]
    const float* sigmas  = (const float*)d_in[2];   // [2048]
    float* out = (float*)d_out;                     // [8192, 2048]

    float* xsq = (float*)d_ws;                      // 8192 f32
    float* csq = xsq + BATCH;                       // 2048 f32
    float* inv = csq + NCENT;                       // 2048 f32

    rbf_norms_kernel<<<BATCH / 8, 256, 0, stream>>>(inputs, BATCH, xsq,
                                                    nullptr, nullptr);
    rbf_norms_kernel<<<NCENT / 8, 256, 0, stream>>>(centers, NCENT, csq,
                                                    sigmas, inv);

    const size_t norm_bytes  = (size_t)(BATCH + 2 * NCENT) * sizeof(float);
    const size_t split_off   = (norm_bytes + 255) & ~(size_t)255;
    const size_t nx          = (size_t)BATCH * NFEAT;
    const size_t nc          = (size_t)NCENT * NFEAT;
    const size_t split_bytes = 2 * (nx + nc) * sizeof(unsigned short);

    dim3 grid(NCENT / 128, BATCH / 128);  // (16, 64)

    if (ws_size >= split_off + split_bytes) {
        unsigned short* xhi = (unsigned short*)((char*)d_ws + split_off);
        unsigned short* xlo = xhi + nx;
        unsigned short* chi = xlo + nx;
        unsigned short* clo = chi + nc;

        rbf_split_kernel<<<(int)(nx / 4 / 256), 256, 0, stream>>>(inputs, xhi, xlo,
                                                                  (int)(nx / 4));
        rbf_split_kernel<<<(int)(nc / 4 / 256), 256, 0, stream>>>(centers, chi, clo,
                                                                  (int)(nc / 4));

        const unsigned shmem = 2 * 4 * TILE_E * sizeof(unsigned short);  // 80 KB
        rbf_main_split<<<grid, 256, shmem, stream>>>(xhi, xlo, chi, clo,
                                                     xsq, csq, inv, out);
    } else {
        rbf_main_f32<<<grid, 256, 0, stream>>>(inputs, centers, xsq, csq, inv, out);
    }
}
